// TaylorVerletFlow_34170759807427
// MI455X (gfx1250) — matrix operand order repaired and checked by reading, not compiled, not run on hardware
//
#include <hip/hip_runtime.h>
#include <hip/hip_bf16.h>

// ---------------------------------------------------------------------------
// TaylorVerletFlow on MI455X (gfx1250). Compute-bound (~144 GFLOP vs ~tens of
// MB of traffic at 23.3 TB/s), so all four time-injection MLPs run as f16
// WMMA GEMMs (v_wmma_f32_16x16x32_f16, f32 accumulate). The concat-t column
// and bias fold into the epilogue as (b[n] + t[i]*wt[n]).
//
// Round-1 fixes vs previous version:
//  * No divergent-looking guards inside the K loop: B-row indices are clamped
//    so loads are always in-bounds; all 4 n-subtiles compute unconditionally;
//    only epilogue stores are predicated. WMMA always runs with full EXEC.
//  * Activations are stored in a WMMA-swizzled K order (swap the middle two
//    8-groups of each 32-chunk), making every A fragment one contiguous 32B
//    v16h load -- B fragments use two 16B v8h loads each.
//  * Unconditional global_prefetch of next k-step weights; unroll-by-2 K loop
//    so next-step loads overlap current-step WMMAs.
// ---------------------------------------------------------------------------

typedef __attribute__((ext_vector_type(16))) _Float16 v16h;
typedef __attribute__((ext_vector_type(8)))  _Float16 v8h;
typedef __attribute__((ext_vector_type(8)))  float    v8f;

#define CLIP_V 20.0f

__device__ __forceinline__ float selu_f(float x) {
    const float sc = 1.0507009873554805f;
    const float al = 1.6732632423543772f;
    return sc * (x > 0.0f ? x : al * (__expf(x) - 1.0f));
}

// Activation K-swizzle: within each 32-chunk, swap 8-groups 1 and 2.
// Slots [0..7,8..15,16..23,24..31] hold original K [0..7,16..23,8..15,24..31],
// so the half-wave A fragment (orig K {0..7,16..23} / {8..15,24..31}) is one
// contiguous 16-element run at offset half*16.
__device__ __forceinline__ int swz32(int p) {
    int g  = (p >> 3) & 3;
    int ng = (g == 1) ? 2 : (g == 2) ? 1 : g;
    return (p & ~31) | (ng << 3) | (p & 7);
}

// --------------------------- weight conversion ------------------------------
// W is (N x (K+1)) f32 row-major; last column is the t-injection weight.
// Emit Wh (N x K) f16 row-major (no swizzle; B layout wants contiguous K)
// plus affine[n] = {b[n], W[n][K]}.
__global__ void convert_weights(const float* __restrict__ W, const float* __restrict__ b,
                                _Float16* __restrict__ Wh, float2* __restrict__ af,
                                int N, int K) {
    int idx = blockIdx.x * blockDim.x + threadIdx.x;
    if (idx >= N * K) return;
    int n = idx / K, k = idx - n * K;
    Wh[idx] = (_Float16)W[(size_t)n * (K + 1) + k];
    if (k == 0) af[n] = make_float2(b[n], W[(size_t)n * (K + 1) + K]);
}

// qp (B x 64) f32 -> q16/p16 (B x 32) f16, stored K-swizzled.
__global__ void convert_inputs(const float* __restrict__ qp,
                               _Float16* __restrict__ q16, _Float16* __restrict__ p16,
                               int B) {
    int idx = blockIdx.x * blockDim.x + threadIdx.x;
    if (idx >= B * 32) return;
    int i = idx >> 5, k = idx & 31;
    int d = (i << 5) | swz32(k);
    q16[d] = (_Float16)qp[(size_t)i * 64 + k];
    p16[d] = (_Float16)qp[(size_t)i * 64 + 32 + k];
}

// ------------------------------ WMMA GEMM -----------------------------------
// C(M x N) = A(M x K, f16, K-swizzled rows) * W(N x K, f16 row-major)^T.
// Epilogue adds b[n] + t[row]*wt[n], then:
//   MODE 0: SELU -> f16 store, K-swizzled columns (feeds next GEMM's A)
//   MODE 1: plain -> f32 store (last layer of q0/p0)
//   MODE 2: clip  -> f16 store (last layer of q1/p1; clipped M matrix)
// Block = 8 waves; wave tile 32x64 (2x4 WMMA frags); block tile 128x128.
template <int MODE>
__global__ __launch_bounds__(256)
void gemm_mlp(const _Float16* __restrict__ A, const _Float16* __restrict__ W,
              const float2* __restrict__ af, const float* __restrict__ t,
              int trow0, void* __restrict__ Y, int M, int N, int K, int ldy) {
    const int lane = threadIdx.x & 31;
    const int wave = threadIdx.x >> 5;
    const int half = lane >> 4;   // K half-split across half-waves
    const int l    = lane & 15;

    const int mbase = blockIdx.x * 128 + (wave & 3) * 32;
    const int nbase = blockIdx.y * 128 + (wave >> 2) * 64;
    if (mbase >= M) return;   // wave-uniform in practice; exits before any WMMA
    if (nbase >= N) return;

    // clamped row indices: loads always in-bounds, stores guarded later
    const int rA0 = (mbase + l      < M) ? (mbase + l)      : (M - 1);
    const int rA1 = (mbase + 16 + l < M) ? (mbase + 16 + l) : (M - 1);
    int nrow[4];
#pragma unroll
    for (int j = 0; j < 4; ++j) {
        int n = nbase + j * 16 + l;
        nrow[j] = (n < N) ? n : (N - 1);
    }

    const _Float16* ap0 = A + (size_t)rA0 * K + half * 16;
    const _Float16* ap1 = A + (size_t)rA1 * K + half * 16;
    const _Float16* bp[4];
#pragma unroll
    for (int j = 0; j < 4; ++j) bp[j] = W + (size_t)nrow[j] * K + half * 8;

    v8f c[2][4];
#pragma unroll
    for (int i = 0; i < 2; ++i)
#pragma unroll
        for (int j = 0; j < 4; ++j) c[i][j] = (v8f){0.f,0.f,0.f,0.f,0.f,0.f,0.f,0.f};

#pragma unroll 2
    for (int kb = 0; kb < K; kb += 32) {
        // A: 32B swizzled fragment loads; B: two 16B loads each (k half*8.. and 16+half*8..)
        v16h a0 = *(const v16h*)(ap0 + kb);
        v16h a1 = *(const v16h*)(ap1 + kb);
        v16h b0 = __builtin_shufflevector(*(const v8h*)(bp[0] + kb), *(const v8h*)(bp[0] + kb + 16), 0,1,2,3,4,5,6,7,8,9,10,11,12,13,14,15);
        v16h b1 = __builtin_shufflevector(*(const v8h*)(bp[1] + kb), *(const v8h*)(bp[1] + kb + 16), 0,1,2,3,4,5,6,7,8,9,10,11,12,13,14,15);
        v16h b2 = __builtin_shufflevector(*(const v8h*)(bp[2] + kb), *(const v8h*)(bp[2] + kb + 16), 0,1,2,3,4,5,6,7,8,9,10,11,12,13,14,15);
        v16h b3 = __builtin_shufflevector(*(const v8h*)(bp[3] + kb), *(const v8h*)(bp[3] + kb + 16), 0,1,2,3,4,5,6,7,8,9,10,11,12,13,14,15);
        // speculative prefetch of next k-step weights (silently dropped if OOB)
        __builtin_prefetch((const void*)(bp[0] + kb + 32), 0, 1);
        __builtin_prefetch((const void*)(bp[1] + kb + 32), 0, 1);
        __builtin_prefetch((const void*)(bp[2] + kb + 32), 0, 1);
        __builtin_prefetch((const void*)(bp[3] + kb + 32), 0, 1);

        c[0][0] = __builtin_amdgcn_wmma_f32_16x16x32_f16(false, a0, false, b0, (short)0, c[0][0], false, false);
        c[1][0] = __builtin_amdgcn_wmma_f32_16x16x32_f16(false, a1, false, b0, (short)0, c[1][0], false, false);
        c[0][1] = __builtin_amdgcn_wmma_f32_16x16x32_f16(false, a0, false, b1, (short)0, c[0][1], false, false);
        c[1][1] = __builtin_amdgcn_wmma_f32_16x16x32_f16(false, a1, false, b1, (short)0, c[1][1], false, false);
        c[0][2] = __builtin_amdgcn_wmma_f32_16x16x32_f16(false, a0, false, b2, (short)0, c[0][2], false, false);
        c[1][2] = __builtin_amdgcn_wmma_f32_16x16x32_f16(false, a1, false, b2, (short)0, c[1][2], false, false);
        c[0][3] = __builtin_amdgcn_wmma_f32_16x16x32_f16(false, a0, false, b3, (short)0, c[0][3], false, false);
        c[1][3] = __builtin_amdgcn_wmma_f32_16x16x32_f16(false, a1, false, b3, (short)0, c[1][3], false, false);
    }

    // hoisted t values for the 16 rows this lane touches
    float tv[2][8];
#pragma unroll
    for (int i = 0; i < 2; ++i)
#pragma unroll
        for (int r = 0; r < 8; ++r) {
            int row = mbase + i * 16 + half * 8 + r;
            if (row >= M) row = M - 1;
            tv[i][r] = t[(size_t)trow0 + row];
        }

#pragma unroll
    for (int j = 0; j < 4; ++j) {
        const int n = nbase + j * 16 + l;
        const bool nv = (n < N);
        const float2 ab = af[nv ? n : (N - 1)];
        const int ncol = (MODE == 0) ? swz32(n) : n;   // hidden layers feed next GEMM
#pragma unroll
        for (int i = 0; i < 2; ++i) {
#pragma unroll
            for (int r = 0; r < 8; ++r) {
                const int row = mbase + i * 16 + half * 8 + r;
                if (!nv || row >= M) continue;
                float y = c[i][j][r] + ab.x + tv[i][r] * ab.y;
                if (MODE == 0) {
                    y = selu_f(y);
                    ((_Float16*)Y)[(size_t)row * ldy + ncol] = (_Float16)y;
                } else if (MODE == 1) {
                    ((float*)Y)[(size_t)row * ldy + ncol] = y;
                } else {
                    y = fminf(fmaxf(y, -CLIP_V), CLIP_V);
                    ((_Float16*)Y)[(size_t)row * ldy + ncol] = (_Float16)y;
                }
            }
        }
    }
}

// ---------------------- batched 32x32 mat-vec + add -------------------------
// out[row0+i, ooff+j] = d0[i,j] + sum_k Mb[i, j*32+k] * qp[row0+i, xoff+k]
__global__ void einsum_add(const float* __restrict__ d0, const _Float16* __restrict__ Mb,
                           const float* __restrict__ qp, int xoff,
                           float* __restrict__ out, int ooff, int row0, int Mc) {
    int idx = blockIdx.x * blockDim.x + threadIdx.x;
    if (idx >= Mc * 32) return;
    int i = idx >> 5, j = idx & 31;
    const v8h*    mr = (const v8h*)(Mb + (size_t)i * 1024 + j * 32);
    const float4* x4 = (const float4*)(qp + (size_t)(row0 + i) * 64 + xoff);
    float s = d0[idx];
#pragma unroll
    for (int v = 0; v < 4; ++v) {
        v8h m = mr[v];
        float4 xa = x4[2 * v], xb = x4[2 * v + 1];
        s += (float)m[0] * xa.x + (float)m[1] * xa.y + (float)m[2] * xa.z + (float)m[3] * xa.w
           + (float)m[4] * xb.x + (float)m[5] * xb.y + (float)m[6] * xb.z + (float)m[7] * xb.w;
    }
    out[(size_t)(row0 + i) * 64 + ooff + j] = s;
}

// ------------------------------ host side -----------------------------------
static inline void launch_gemm(int mode, dim3 g, const _Float16* A, const _Float16* W,
                               const float2* af, const float* t, int trow0, void* Y,
                               int M, int N, int K, int ldy, hipStream_t s) {
    if (mode == 0)      gemm_mlp<0><<<g, 256, 0, s>>>(A, W, af, t, trow0, Y, M, N, K, ldy);
    else if (mode == 1) gemm_mlp<1><<<g, 256, 0, s>>>(A, W, af, t, trow0, Y, M, N, K, ldy);
    else                gemm_mlp<2><<<g, 256, 0, s>>>(A, W, af, t, trow0, Y, M, N, K, ldy);
}

extern "C" void kernel_launch(void* const* d_in, const int* in_sizes, int n_in,
                              void* d_out, int out_size, void* d_ws, size_t ws_size,
                              hipStream_t stream) {
    const float* qp = (const float*)d_in[0];
    const float* t  = (const float*)d_in[1];
    const int B = in_sizes[0] / 64;

    // nets: [q0, p0, q1, p1]; each contributes (W0,b0,W1,b1,W2,b2) in d_in order
    const float* P[4][6];
    int outN[4] = {32, 32, 1024, 1024};
    for (int s = 0; s < 4; ++s)
        for (int j = 0; j < 6; ++j)
            P[s][j] = (const float*)d_in[2 + s * 6 + j];

    // workspace bump allocator (256B aligned); total ~42 MB for CH=8192
    char* wsp = (char*)d_ws;
    size_t off = 0;
    auto alloc = [&](size_t bytes) -> void* {
        void* p = wsp + off;
        off += (bytes + 255) & ~(size_t)255;
        return p;
    };

    _Float16 *W0h[4], *W1h[4], *W2h[4];
    float2   *af0[4], *af1[4], *af2[4];
    for (int s = 0; s < 4; ++s) {
        W0h[s] = (_Float16*)alloc((size_t)512 * 32 * 2);
        W1h[s] = (_Float16*)alloc((size_t)512 * 512 * 2);
        W2h[s] = (_Float16*)alloc((size_t)outN[s] * 512 * 2);
        af0[s] = (float2*)alloc((size_t)512 * 8);
        af1[s] = (float2*)alloc((size_t)512 * 8);
        af2[s] = (float2*)alloc((size_t)outN[s] * 8);
    }
    _Float16* q16 = (_Float16*)alloc((size_t)B * 32 * 2);
    _Float16* p16 = (_Float16*)alloc((size_t)B * 32 * 2);

    int CH = 8192;
    if (B % CH != 0) CH = B;   // fall back to single chunk for odd sizes
    _Float16* H0   = (_Float16*)alloc((size_t)CH * 512 * 2);
    _Float16* H1   = (_Float16*)alloc((size_t)CH * 512 * 2);
    float*    d0b  = (float*)alloc((size_t)CH * 32 * 4);
    _Float16* Mbuf = (_Float16*)alloc((size_t)CH * 1024 * 2);
    float* out = (float*)d_out;

    // --- one-time (per launch) conversions ---
    for (int s = 0; s < 4; ++s) {
        convert_weights<<<(512 * 32 + 255) / 256, 256, 0, stream>>>(P[s][0], P[s][1], W0h[s], af0[s], 512, 32);
        convert_weights<<<(512 * 512 + 255) / 256, 256, 0, stream>>>(P[s][2], P[s][3], W1h[s], af1[s], 512, 512);
        convert_weights<<<(outN[s] * 512 + 255) / 256, 256, 0, stream>>>(P[s][4], P[s][5], W2h[s], af2[s], outN[s], 512);
    }
    convert_inputs<<<(B * 32 + 255) / 256, 256, 0, stream>>>(qp, q16, p16, B);

    // --- chunked MLP pipeline ---
    for (int i0 = 0; i0 < B; i0 += CH) {
        const int Mc = (B - i0 < CH) ? (B - i0) : CH;
        dim3 gH((Mc + 127) / 128, (512 + 127) / 128);

        auto run_net = [&](int s, const _Float16* xin, int mode2, void* y2, int ldy2) {
            launch_gemm(0, gH, xin, W0h[s], af0[s], t, i0, H0, Mc, 512, 32, 512, stream);
            launch_gemm(0, gH, H0, W1h[s], af1[s], t, i0, H1, Mc, 512, 512, 512, stream);
            dim3 g2((Mc + 127) / 128, (outN[s] + 127) / 128);
            launch_gemm(mode2, g2, H1, W2h[s], af2[s], t, i0, y2, Mc, outN[s], 512, ldy2, stream);
        };

        const _Float16* xp = p16 + (size_t)i0 * 32;  // q-side nets eat p
        const _Float16* xq = q16 + (size_t)i0 * 32;  // p-side nets eat q

        // dq = mlp_q0(p,t) + clip(mlp_q1(p,t)) @ q
        run_net(0, xp, 1, d0b, 32);
        run_net(2, xp, 2, Mbuf, 1024);
        einsum_add<<<(Mc * 32 + 255) / 256, 256, 0, stream>>>(d0b, Mbuf, qp, 0, out, 0, i0, Mc);

        // dp = mlp_p0(q,t) + clip(mlp_p1(q,t)) @ p
        run_net(1, xq, 1, d0b, 32);
        run_net(3, xq, 2, Mbuf, 1024);
        einsum_add<<<(Mc * 32 + 255) / 256, 256, 0, stream>>>(d0b, Mbuf, qp, 32, out, 32, i0, Mc);
    }
}
